// MultiHeadAttention_88734024335731
// MI455X (gfx1250) — matrix operand order repaired and checked by reading, not compiled, not run on hardware
//
#include <hip/hip_runtime.h>
#include <cstdint>
#include <cstddef>

// ---------------------------------------------------------------------------
// MHA forward for MI455X (gfx1250, wave32, WMMA + TDM).
// B=2, S=2048, D=2048, H=16, HD=128.
// Pipeline: cvt(f32->bf16) -> 3x GEMM (Q,K,V) -> flash-attention -> GEMM+bias.
// Matmuls: v_wmma_f32_16x16x32_bf16.  Tile staging: tensor_load_to_lds (TDM,
// TENSORcnt) when the builtin exists, else global->VGPR->ds_store fallback.
// Workspace usage (~118 MB): xb, wqb, wkb, wvb, wob, qb, kb, vb, aob (bf16).
// ---------------------------------------------------------------------------

typedef __attribute__((ext_vector_type(16))) __bf16 v16bf;
typedef __attribute__((ext_vector_type(8)))  float  v8f;

#if defined(__HIP_DEVICE_COMPILE__) && defined(__has_builtin)
#if __has_builtin(__builtin_amdgcn_tensor_load_to_lds) && \
    __has_builtin(__builtin_amdgcn_s_wait_tensorcnt)
#define CDNA5_TDM 1
#endif
#endif
#ifndef CDNA5_TDM
#define CDNA5_TDM 0
#endif

union FragU { uint4 u[2]; v16bf v; };
union Pack8 { uint4 u; __bf16 e[8]; };

static __device__ __forceinline__ v8f wmma_bf16(v16bf a, v16bf b, v8f c) {
  // (neg_a, A, neg_b, B, c_mod, C, reuse_a, reuse_b)
  return __builtin_amdgcn_wmma_f32_16x16x32_bf16(false, a, false, b, (short)0, c,
                                                 false, false);
}

// A-fragment: 16x32 bf16, lane = M row; lane<16 K=k0..k0+15,
// lane>=16 K=+16 (matches B-fragment order).
static __device__ __forceinline__ v16bf load_fragA(const __bf16* t, int m0, int k0,
                                                   int ld, int lane) {
  const __bf16* p = t + (size_t)(m0 + (lane & 15)) * ld + k0 + ((lane >> 4) & 1) * 16;
  FragU f;
  f.u[0] = *(const uint4*)p;
  f.u[1] = *(const uint4*)(p + 8);
  return f.v;
}

// B-fragment: 32x16 bf16, lane = N col; lane<16 K=k0..k0+15, lane>=16 K=+16.
// Tile stored [N][K] row-major (K contiguous) => one 32-byte run per lane.
static __device__ __forceinline__ v16bf load_fragB(const __bf16* t, int n0, int k0,
                                                   int ld, int lane) {
  const __bf16* p = t + (size_t)(n0 + (lane & 15)) * ld + k0 + ((lane >> 4) & 1) * 16;
  FragU f;
  f.u[0] = *(const uint4*)p;
  f.u[1] = *(const uint4*)(p + 8);
  return f.v;
}

// ---------------------------------------------------------------------------
// TDM 2D tile load: global (row-major, stride_elems bf16 per row) -> LDS with
// row padding (pad_amount DWORDs inserted after every 2^(pad_interval+3) bytes,
// matching our padded LDS strides).  Descriptor per CDNA5 ISA Ch.8:
//   group0: count=1 | lds_addr | global_addr[56:0] | type=2
//   group1: data_size=1(2B) pad_enable pad codes | tensor dims | tile dims |
//           tensor_dim0_stride
// Groups 2/3 (and the trailing int32x8 of the 6-arg builtin form) are unused
// for 2D tiles and passed as zeros.  Issued by one wave (EXEC ignored by TDM);
// completion via s_wait_tensorcnt.
// ---------------------------------------------------------------------------
static __device__ __forceinline__ void tdm_load_2d(const __bf16* g, __bf16* l,
                                                   uint32_t tile_d0, uint32_t tile_d1,
                                                   uint64_t stride_elems,
                                                   uint32_t pad_interval_code,
                                                   uint32_t pad_amount_code) {
#if CDNA5_TDM
  typedef unsigned int u32x4 __attribute__((ext_vector_type(4)));
  typedef int i32x8 __attribute__((ext_vector_type(8)));
  typedef int i32x4 __attribute__((ext_vector_type(4)));

  uint64_t ga = (uint64_t)(uintptr_t)g;            // global VA (57-bit)
  uint32_t la = (uint32_t)(uintptr_t)l;            // low 32 bits = LDS byte offset

  u32x4 g0 = { 1u,                                              // count=1, user D#
               la,                                              // lds_addr
               (uint32_t)ga,                                    // global_addr[31:0]
               (uint32_t)((ga >> 32) & 0x01FFFFFFu) | (2u << 30) }; // [56:32]|type=2

  uint32_t w[8] = {};
  w[0] = (1u << 16)                      // data_size = 1 -> 2-byte elements
       | (1u << 20)                      // pad_enable
       | (pad_interval_code << 22)       // pad every 2^(code+1) DWORDs
       | (pad_amount_code << 25);        // insert (code+1) DWORDs
  // tensor_dim0 at bit48, tensor_dim1 at bit80 (tensor == tile: no OOB clip)
  w[1] = (tile_d0 & 0xFFFFu) << 16;
  w[2] = (tile_d0 >> 16) | ((tile_d1 & 0xFFFFu) << 16);
  w[3] = (tile_d1 >> 16) | (tile_d0 << 16);          // tile_dim0 at bit112
  w[4] = tile_d1;                                    // tile_dim1 (tile_dim2=0)
  w[5] = (uint32_t)stride_elems;                     // tensor_dim0_stride at bit160
  w[6] = (uint32_t)((stride_elems >> 32) & 0xFFFFu); // stride[47:32]; dim1_stride=0
  w[7] = 0;
  i32x8 g1;
#pragma unroll
  for (int i = 0; i < 8; ++i) g1[i] = (int)w[i];
  i32x4 z4 = { 0, 0, 0, 0 };
  i32x8 z8 = { 0, 0, 0, 0, 0, 0, 0, 0 };
  __builtin_amdgcn_tensor_load_to_lds(g0, g1, z4, z4, z8, 0);
#else
  (void)g; (void)l; (void)tile_d0; (void)tile_d1; (void)stride_elems;
  (void)pad_interval_code; (void)pad_amount_code;
#endif
}

static __device__ __forceinline__ void tdm_wait_all() {
#if CDNA5_TDM
  __builtin_amdgcn_s_wait_tensorcnt(0);
#endif
}

// ---------------------------------------------------------------------------
// fp32 -> bf16 conversion, 8 elements / thread, vectorized 16B loads/stores.
// ---------------------------------------------------------------------------
__global__ __launch_bounds__(256) void cvt_f32_bf16(const float* __restrict__ in,
                                                    __bf16* __restrict__ out, int n) {
  int i = (blockIdx.x * 256 + threadIdx.x) * 8;
  if (i + 8 <= n) {
    float4 a = *(const float4*)(in + i);
    float4 b = *(const float4*)(in + i + 4);
    Pack8 w;
    w.e[0] = (__bf16)a.x; w.e[1] = (__bf16)a.y; w.e[2] = (__bf16)a.z; w.e[3] = (__bf16)a.w;
    w.e[4] = (__bf16)b.x; w.e[5] = (__bf16)b.y; w.e[6] = (__bf16)b.z; w.e[7] = (__bf16)b.w;
    *(uint4*)(out + i) = w.u;
  }
}

// ---------------------------------------------------------------------------
// C[M,N] = A[M,K] @ W[N,K]^T  (NT layout; both operands K-contiguous bf16).
// 128x128 block tile, K-tile 64, double-buffered LDS staged by the TDM:
// next tile's DMA is issued right after the top barrier and lands under the
// 16 WMMAs/wave of the current tile.  8 waves (2x4), each wave 64x32.
// F32OUT=false: bf16 C (Q/K/V). F32OUT=true: fp32 C + bias (output proj).
// Dynamic LDS: 2*(A+B) padded tiles = 73,728 B.
// ---------------------------------------------------------------------------
template <bool F32OUT>
__global__ __launch_bounds__(256) void gemm_nt(const __bf16* __restrict__ A,
                                               const __bf16* __restrict__ W,
                                               __bf16* __restrict__ outB,
                                               float* __restrict__ outF,
                                               const float* __restrict__ bias,
                                               int M, int N, int K) {
  constexpr int LDK  = 72;           // 64 + 8 pad (16B-aligned, spreads banks)
  constexpr int TILE = 128 * LDK;
  extern __shared__ __align__(16) char gsm[];
  __bf16* sA = (__bf16*)gsm;         // [2][128*LDK]
  __bf16* sB = sA + 2 * TILE;        // [2][128*LDK]

  const int tid  = threadIdx.x;
  const int lane = tid & 31;
  const int wave = tid >> 5;
  const int wm = (wave >> 2) * 64;   // 0 or 64
  const int wn = (wave & 3) * 32;    // 0,32,64,96
  const int rowBase = blockIdx.y * 128;
  const int colBase = blockIdx.x * 128;

  v8f acc[4][2] = {};

  auto stage = [&](int buf, int k0) {
#if CDNA5_TDM
    if (wave == 0) {
      // 128 rows x 64 bf16; LDS rows padded 128B -> 144B (pad 4 DW per 32 DW)
      tdm_load_2d(A + (size_t)rowBase * K + k0, sA + buf * TILE, 64, 128,
                  (uint64_t)K, 4, 3);
      tdm_load_2d(W + (size_t)colBase * K + k0, sB + buf * TILE, 64, 128,
                  (uint64_t)K, 4, 3);
    }
#else
#pragma unroll
    for (int i = 0; i < 4; ++i) {
      int c = tid + (i << 8);
      int r = c >> 3, cc = (c & 7) << 3;
      *(uint4*)&sA[buf * TILE + r * LDK + cc] =
          *(const uint4*)(A + (size_t)(rowBase + r) * K + k0 + cc);
      *(uint4*)&sB[buf * TILE + r * LDK + cc] =
          *(const uint4*)(W + (size_t)(colBase + r) * K + k0 + cc);
    }
    if (k0 + 64 < K) {  // hint next K tile into cache -> global_prefetch_b8
      __builtin_prefetch(A + (size_t)(rowBase + (tid >> 1)) * K + k0 + 64 + (tid & 1) * 32, 0, 1);
      __builtin_prefetch(W + (size_t)(colBase + (tid >> 1)) * K + k0 + 64 + (tid & 1) * 32, 0, 1);
    }
#endif
  };

  stage(0, 0);
  int cur = 0;
  for (int k0 = 0; k0 < K; k0 += 64) {
    if (wave == 0) tdm_wait_all();   // staged tile landed in LDS
    __syncthreads();                 // ...and is visible to all waves
    if (k0 + 64 < K) stage(cur ^ 1, k0 + 64);  // DMA next tile under compute

    const __bf16* tA = sA + cur * TILE;
    const __bf16* tB = sB + cur * TILE;
#pragma unroll
    for (int kk = 0; kk < 64; kk += 32) {
      v16bf b0 = load_fragB(tB, wn + 0,  kk, LDK, lane);
      v16bf b1 = load_fragB(tB, wn + 16, kk, LDK, lane);
#pragma unroll
      for (int mt = 0; mt < 4; ++mt) {
        v16bf a0 = load_fragA(tA, wm + mt * 16, kk, LDK, lane);
        acc[mt][0] = wmma_bf16(a0, b0, acc[mt][0]);
        acc[mt][1] = wmma_bf16(a0, b1, acc[mt][1]);
      }
    }
    cur ^= 1;
    __syncthreads();  // all fragment reads done before this buffer is restaged
  }

  // Epilogue. C layout: VGPR r -> row r (lanes 0-15) / row r+8 (lanes 16-31).
  const int hi8 = (lane >> 4) << 3;
  const int cl  = lane & 15;
#pragma unroll
  for (int mt = 0; mt < 4; ++mt)
#pragma unroll
    for (int nt = 0; nt < 2; ++nt)
#pragma unroll
      for (int r = 0; r < 8; ++r) {
        int row = rowBase + wm + mt * 16 + r + hi8;
        int col = colBase + wn + nt * 16 + cl;
        float v = acc[mt][nt][r];
        if (F32OUT)
          outF[(size_t)row * N + col] = v + bias[col];
        else
          outB[(size_t)row * N + col] = (__bf16)v;
      }
}

// ---------------------------------------------------------------------------
// Flash attention: one block per (128-query tile, b*H+h). 8 waves, each owns
// 16 query rows. Online softmax; P re-fragmented through LDS (reuses K tile).
// Q/K tiles staged by TDM (pad 256B rows -> 272B); V transposed via registers.
// Q/K/V are bf16 [B,S,D] with head h occupying columns h*HD .. h*HD+127.
// ---------------------------------------------------------------------------
__global__ __launch_bounds__(256) void attn_kernel(const __bf16* __restrict__ qg,
                                                   const __bf16* __restrict__ kg,
                                                   const __bf16* __restrict__ vg,
                                                   __bf16* __restrict__ og) {
  constexpr int S = 2048, D = 2048, H = 16;
  constexpr int LDT = 136;  // 128 + 8 pad
  constexpr float SCALE = 0.08838834764831845f;  // 1/sqrt(128)

  extern __shared__ __align__(16) char smem[];
  __bf16* Qs = (__bf16*)smem;          // [128][136]
  __bf16* Ks = Qs + 128 * LDT;         // [128][136]  (kv-major)
  __bf16* Vt = Ks + 128 * LDT;         // [128][136]  transposed: [d][kv]
  __bf16* Ps = Ks;                     // P reuses the K buffer

  const int tid  = threadIdx.x;
  const int lane = tid & 31;
  const int wave = tid >> 5;
  const int qt = blockIdx.x;           // query tile (0..15)
  const int b  = blockIdx.y >> 4;
  const int h  = blockIdx.y & (H - 1);
  const int hi8 = (lane >> 4) << 3;
  const int cl  = lane & 15;
  const int wq0 = wave * 16;
  const float NEG_INF = -__builtin_inff();

  const __bf16* qp = qg + ((size_t)b * S) * D + (size_t)h * 128;
  const __bf16* kp = kg + ((size_t)b * S) * D + (size_t)h * 128;
  const __bf16* vp = vg + ((size_t)b * S) * D + (size_t)h * 128;

  // Stage Q tile (128x128 bf16).
#if CDNA5_TDM
  if (wave == 0)
    tdm_load_2d(qp + (size_t)(qt * 128) * D, Qs, 128, 128, (uint64_t)D, 5, 3);
#else
#pragma unroll
  for (int i = 0; i < 8; ++i) {
    int c = tid + (i << 8);
    int r = c >> 4, d0 = (c & 15) << 3;
    *(uint4*)&Qs[r * LDT + d0] = *(const uint4*)(qp + (size_t)(qt * 128 + r) * D + d0);
  }
#endif

  float mi[8], li[8];
#pragma unroll
  for (int r = 0; r < 8; ++r) { mi[r] = NEG_INF; li[r] = 0.0f; }
  v8f o[8] = {};

  for (int kt = 0; kt <= qt; ++kt) {
    __syncthreads();  // previous tile's Ps/Vt reads complete
#if CDNA5_TDM
    if (wave == 0)
      tdm_load_2d(kp + (size_t)(kt * 128) * D, Ks, 128, 128, (uint64_t)D, 5, 3);
#endif
    // Stage V transposed (and K synchronously when no TDM).
#pragma unroll
    for (int i = 0; i < 8; ++i) {
      int c = tid + (i << 8);
      int kv = c >> 4, d0 = (c & 15) << 3;
#if !CDNA5_TDM
      *(uint4*)&Ks[kv * LDT + d0] = *(const uint4*)(kp + (size_t)(kt * 128 + kv) * D + d0);
#endif
      Pack8 w;
      w.u = *(const uint4*)(vp + (size_t)(kt * 128 + kv) * D + d0);
#pragma unroll
      for (int j = 0; j < 8; ++j) Vt[(d0 + j) * LDT + kv] = w.e[j];
    }
    if (wave == 0) tdm_wait_all();
    __syncthreads();

    // S = Q @ K^T for this wave's 16 rows x 128 keys.
    v8f s[8] = {};
#pragma unroll
    for (int ks = 0; ks < 128; ks += 32) {
      v16bf aq = load_fragA(Qs, wq0, ks, LDT, lane);
#pragma unroll
      for (int nt = 0; nt < 8; ++nt) {
        v16bf bk = load_fragB(Ks, nt * 16, ks, LDT, lane);
        s[nt] = wmma_bf16(aq, bk, s[nt]);
      }
    }

    // Scale + causal mask + online softmax.
    const bool diag = (kt == qt);
    float tmax[8];
#pragma unroll
    for (int r = 0; r < 8; ++r) tmax[r] = NEG_INF;
#pragma unroll
    for (int nt = 0; nt < 8; ++nt)
#pragma unroll
      for (int r = 0; r < 8; ++r) {
        float v = s[nt][r] * SCALE;
        if (diag) {
          int row = qt * 128 + wq0 + r + hi8;
          int col = kt * 128 + nt * 16 + cl;
          if (col > row) v = NEG_INF;
        }
        s[nt][r] = v;
        tmax[r] = fmaxf(tmax[r], v);
      }
#pragma unroll
    for (int r = 0; r < 8; ++r)
#pragma unroll
      for (int off = 8; off >= 1; off >>= 1)
        tmax[r] = fmaxf(tmax[r], __shfl_xor(tmax[r], off, 32));

    float osc[8], rs[8];
#pragma unroll
    for (int r = 0; r < 8; ++r) {
      float mn = fmaxf(mi[r], tmax[r]);
      osc[r] = __expf(mi[r] - mn);
      mi[r] = mn;
      li[r] *= osc[r];
      rs[r] = 0.0f;
    }
#pragma unroll
    for (int nt = 0; nt < 8; ++nt)
#pragma unroll
      for (int r = 0; r < 8; ++r) {
        float p = __expf(s[nt][r] - mi[r]);
        s[nt][r] = p;
        rs[r] += p;
      }
#pragma unroll
    for (int r = 0; r < 8; ++r) {
#pragma unroll
      for (int off = 8; off >= 1; off >>= 1)
        rs[r] += __shfl_xor(rs[r], off, 32);
      li[r] += rs[r];
    }
#pragma unroll
    for (int nt = 0; nt < 8; ++nt)
#pragma unroll
      for (int r = 0; r < 8; ++r) o[nt][r] *= osc[r];

    __syncthreads();  // all waves done reading Ks before P overwrites it
#pragma unroll
    for (int nt = 0; nt < 8; ++nt)
#pragma unroll
      for (int r = 0; r < 8; ++r)
        Ps[(wq0 + r + hi8) * LDT + nt * 16 + cl] = (__bf16)s[nt][r];
    __syncthreads();

    // O += P @ V  (Vt is [d][kv], so B-fragments are contiguous runs).
#pragma unroll
    for (int ks = 0; ks < 128; ks += 32) {
      v16bf ap = load_fragA(Ps, wq0, ks, LDT, lane);
#pragma unroll
      for (int nt = 0; nt < 8; ++nt) {
        v16bf bv = load_fragB(Vt, nt * 16, ks, LDT, lane);
        o[nt] = wmma_bf16(ap, bv, o[nt]);
      }
    }
  }

  // Normalize and write bf16 output in [B,S,D] (head-merged) layout.
  __bf16* op = og + ((size_t)b * S) * D + (size_t)h * 128;
  float inv[8];
#pragma unroll
  for (int r = 0; r < 8; ++r) inv[r] = 1.0f / li[r];
#pragma unroll
  for (int nt = 0; nt < 8; ++nt)
#pragma unroll
    for (int r = 0; r < 8; ++r) {
      int row = qt * 128 + wq0 + r + hi8;
      op[(size_t)row * D + nt * 16 + cl] = (__bf16)(o[nt][r] * inv[r]);
    }
}

// ---------------------------------------------------------------------------
extern "C" void kernel_launch(void* const* d_in, const int* in_sizes, int n_in,
                              void* d_out, int out_size, void* d_ws, size_t ws_size,
                              hipStream_t stream) {
  (void)in_sizes; (void)n_in; (void)out_size; (void)ws_size;
  constexpr int B = 2, S = 2048, D = 2048;
  constexpr int M = B * S;                 // 4096
  constexpr size_t XN = (size_t)M * D;     // 8,388,608
  constexpr size_t WN = (size_t)D * D;     // 4,194,304

  const float* x  = (const float*)d_in[0];
  const float* wq = (const float*)d_in[1];
  const float* wk = (const float*)d_in[2];
  const float* wv = (const float*)d_in[3];
  const float* wo = (const float*)d_in[4];
  const float* bo = (const float*)d_in[5];
  float* out = (float*)d_out;

  char* ws = (char*)d_ws;
  __bf16* xb  = (__bf16*)ws;                 ws += XN * 2;
  __bf16* wqb = (__bf16*)ws;                 ws += WN * 2;
  __bf16* wkb = (__bf16*)ws;                 ws += WN * 2;
  __bf16* wvb = (__bf16*)ws;                 ws += WN * 2;
  __bf16* wob = (__bf16*)ws;                 ws += WN * 2;
  __bf16* qb  = (__bf16*)ws;                 ws += XN * 2;
  __bf16* kb  = (__bf16*)ws;                 ws += XN * 2;
  __bf16* vb  = (__bf16*)ws;                 ws += XN * 2;
  __bf16* aob = (__bf16*)ws;                 // + XN*2  (total ~118 MB)

  // 1) fp32 -> bf16
  cvt_f32_bf16<<<(int)(XN / 2048), 256, 0, stream>>>(x,  xb,  (int)XN);
  cvt_f32_bf16<<<(int)(WN / 2048), 256, 0, stream>>>(wq, wqb, (int)WN);
  cvt_f32_bf16<<<(int)(WN / 2048), 256, 0, stream>>>(wk, wkb, (int)WN);
  cvt_f32_bf16<<<(int)(WN / 2048), 256, 0, stream>>>(wv, wvb, (int)WN);
  cvt_f32_bf16<<<(int)(WN / 2048), 256, 0, stream>>>(wo, wob, (int)WN);

  // 2) Q/K/V projections: [4096,2048] = x @ W^T   (73,728 B dynamic LDS)
  dim3 gg(D / 128, M / 128);  // (16, 32)
  constexpr int GEMM_LDS = 4 * 128 * 72 * 2;
  gemm_nt<false><<<gg, 256, GEMM_LDS, stream>>>(xb, wqb, qb, nullptr, nullptr, M, D, D);
  gemm_nt<false><<<gg, 256, GEMM_LDS, stream>>>(xb, wkb, kb, nullptr, nullptr, M, D, D);
  gemm_nt<false><<<gg, 256, GEMM_LDS, stream>>>(xb, wvb, vb, nullptr, nullptr, M, D, D);

  // 3) Flash attention per (q-tile, b*H+h); 102 KB dynamic LDS per block.
  dim3 ga(S / 128, B * 16);   // (16, 32)
  attn_kernel<<<ga, 256, 3 * 128 * 136 * 2, stream>>>(qb, kb, vb, aob);

  // 4) Output projection + bias (fp32 out).
  gemm_nt<true><<<gg, 256, GEMM_LDS, stream>>>(aob, wob, nullptr, out, bo, M, D, D);
}
